// GraphConditionedTwoWayBlock_87952340287722
// MI455X (gfx1250) — matrix operand order repaired and checked by reading, not compiled, not run on hardware
//
#include <hip/hip_runtime.h>

// GraphConditionedTwoWayBlock for MI455X (gfx1250): all GEMMs + attention
// matmuls run on v_wmma_f32_16x16x32_bf16 (wave32), fp32 accumulate,
// LN/softmax/residual in fp32. GEMM tiles are fed by the Tensor Data Mover
// (tensor_load_to_lds, TENSORcnt) when the builtin is available.

typedef float  v8f   __attribute__((ext_vector_type(8)));
typedef __bf16 v16bf __attribute__((ext_vector_type(16)));
typedef __bf16 v8bf  __attribute__((ext_vector_type(8)));

// ---- Tensor Data Mover (TDM) support ---------------------------------------

#if defined(__has_builtin)
#if __has_builtin(__builtin_amdgcn_tensor_load_to_lds)
#define GCTB_HAVE_TDM 1
#endif
#endif

#ifdef GCTB_HAVE_TDM
typedef unsigned int gctb_u32x4 __attribute__((ext_vector_type(4)));
typedef int          gctb_i32x8 __attribute__((ext_vector_type(8)));
typedef int          gctb_i32x4 __attribute__((ext_vector_type(4)));

// 2-D tile DMA: rows x 32 elements of 2-byte data, row stride = K elements,
// from global tile base `gsrc` into LDS byte offset `lds_off`.
// D# per CDNA5 ISA 8.3/8.4: group0 = {count/type, lds_addr, global_addr},
// group1 = {data_size, tensor dims, tile dims, dim0 stride}.
// This toolchain exposes the 6-arg builtin (clang-23 / therock-10.0 form):
//   (uint32x4 g0, int32x8 g1, int32x4 g2, int32x4 g3, int32x8 extra, i32 cpol)
__device__ __forceinline__ void gctb_tdm_tile_2d(unsigned lds_off,
                                                 const void* gsrc,
                                                 unsigned K, unsigned rows,
                                                 unsigned tensor_rows) {
  const unsigned long long ga = (unsigned long long)gsrc;
  gctb_u32x4 g0;
  g0[0] = 1u;                                        // count=1, user descriptor
  g0[1] = lds_off;                                   // lds_addr (bytes)
  g0[2] = (unsigned)ga;                              // global_addr[31:0]
  g0[3] = (unsigned)((ga >> 32) & 0x01ffffffu) | (2u << 30);  // [56:32] | type=2
  gctb_i32x8 g1;
  g1[0] = 0x00010000;                                // data_size=1 (2 bytes)
  g1[1] = (int)((K & 0xffffu) << 16);                // tensor_dim0[15:0]
  g1[2] = (int)((K >> 16) | ((tensor_rows & 0xffffu) << 16));  // dim0 hi | dim1 lo
  g1[3] = (int)(((tensor_rows >> 16) & 0xffffu) | (32u << 16)); // dim1 hi | tile_dim0=32
  g1[4] = (int)rows;                                 // tile_dim1, tile_dim2=0
  g1[5] = (int)K;                                    // tensor_dim0_stride[31:0]
  g1[6] = 0;
  g1[7] = 0;
  const gctb_i32x4 gz4 = {0, 0, 0, 0};
  const gctb_i32x8 gz8 = {0, 0, 0, 0, 0, 0, 0, 0};
  __builtin_amdgcn_tensor_load_to_lds(g0, g1, gz4, gz4, gz8, 0);
}

__device__ __forceinline__ void gctb_wait_tensor() {
#if __has_builtin(__builtin_amdgcn_s_wait_tensorcnt)
  __builtin_amdgcn_s_wait_tensorcnt(0);
#else
  asm volatile("s_wait_tensorcnt 0x0" ::: "memory");
#endif
}
#endif  // GCTB_HAVE_TDM

// The TDM writes LDS behind the compiler's back; escape the tile pointers with
// a memory clobber so the subsequent ds_loads are not folded to undef.
#define GCTB_LDS_ESCAPE1(p0)         asm volatile("" : : "r"(p0) : "memory")
#define GCTB_LDS_ESCAPE2(p0, p1)     asm volatile("" : : "r"(p0), "r"(p1) : "memory")
#define GCTB_LDS_ESCAPE3(p0, p1, p2) asm volatile("" : : "r"(p0), "r"(p1), "r"(p2) : "memory")

// ---- WMMA helpers -----------------------------------------------------------

__device__ __forceinline__ v8f wmma_bf16(v16bf a, v16bf b, v8f c) {
  return __builtin_amdgcn_wmma_f32_16x16x32_bf16(
      /*neg_a=*/false, a, /*neg_b=*/false, b,
      /*c_mod=*/(short)0, c, /*reuse_a=*/false, /*reuse_b=*/false);
}

// A-matrix 16x32 bf16 fragment (same K order as bfrag_bf): lane l holds row M=l%16;
// elements i -> K = (l/16)*16 + i (contiguous along K)
__device__ __forceinline__ v16bf afrag_bf(const __bf16* rowp, int lane) {
  const __bf16* p = rowp + ((lane >> 4) << 4);
  v16bf f;
#pragma unroll
  for (int i = 0; i < 8; ++i) { f[i] = p[i]; f[i + 8] = p[i + 8]; }
  return f;
}

// Same fragment but sourced from fp32 (softmax probabilities in LDS).
__device__ __forceinline__ v16bf afrag_f32(const float* rowp, int lane) {
  const float* p = rowp + ((lane >> 4) << 4);
  v16bf f;
#pragma unroll
  for (int i = 0; i < 8; ++i) { f[i] = (__bf16)p[i]; f[i + 8] = (__bf16)p[i + 8]; }
  return f;
}

// B-matrix 32x16 bf16 fragment: lane l holds col N=l%16;
// elements i -> K = (l/16)*16 + i (contiguous along K).
__device__ __forceinline__ v16bf bfrag_bf(const __bf16* colp, int lane) {
  const __bf16* p = colp + ((lane >> 4) << 4);
  v16bf f;
#pragma unroll
  for (int i = 0; i < 16; ++i) f[i] = p[i];
  return f;
}

// ---- weight fp32 -> bf16 ----------------------------------------------------

__global__ __launch_bounds__(256) void gctb_cvtw(const float* __restrict__ in,
                                                 __bf16* __restrict__ out, int n) {
  int i = blockIdx.x * 256 + threadIdx.x;
  if (i < n) out[i] = (__bf16)in[i];
}

// ---- LayerNorm over D=256, output bf16 (compact rows) -----------------------
// Input token index is remapped: t_in = (r/chunk)*stride + r%chunk + off.

__global__ __launch_bounds__(256) void gctb_ln(const float* __restrict__ in,
                                               const float* __restrict__ gam,
                                               const float* __restrict__ bet,
                                               __bf16* __restrict__ out,
                                               long chunk, long stride, long off) {
  __shared__ float red[256];
  const long r = blockIdx.x;
  const long t = (r / chunk) * stride + (r % chunk) + off;
  const int d = threadIdx.x;
  const float x = in[t * 256 + d];
  red[d] = x; __syncthreads();
#pragma unroll
  for (int s = 128; s > 0; s >>= 1) { if (d < s) red[d] += red[d + s]; __syncthreads(); }
  const float mean = red[0] * (1.f / 256.f);
  __syncthreads();
  const float dx = x - mean;
  red[d] = dx * dx; __syncthreads();
#pragma unroll
  for (int s = 128; s > 0; s >>= 1) { if (d < s) red[d] += red[d + s]; __syncthreads(); }
  const float var = red[0] * (1.f / 256.f);
  const float y = dx * rsqrtf(var + 1e-5f) * gam[d] + bet[d];
  out[r * 256 + d] = (__bf16)y;
}

// ---- Tiled WMMA GEMM: C[m,n] = sum_k A[m,k]*W[n,k] + bias[n] ----------------
// 128x128 block tile, K-step 32, 8 waves: wave (wm,wn) owns 64x32 (4x2 WMMA).
// Tiles are DMA'd into LDS with tensor_load_to_lds when available.
// mode 0: write bf16 to outb (stride outStride).
// mode 1: outf[idx] = res[idx] + C, idx via token remap (chunk,stride,off).

__global__ __launch_bounds__(256) void gctb_gemm(
    const __bf16* __restrict__ A, const __bf16* __restrict__ W,
    const float* __restrict__ bias, const float* __restrict__ res,
    float* __restrict__ outf, __bf16* __restrict__ outb,
    int K, int outStride, long mapChunk, long mapStride, long mapOff, int mode) {
  // LDS layout (declaration order): At @ 0, Bt @ 8192  (used by TDM descriptors)
  __shared__ __bf16 At[128][32];
  __shared__ __bf16 Bt[128][32];
  const int tid = threadIdx.x, lane = tid & 31, w = tid >> 5;
  const int wm = w & 1, wn = w >> 1;
  const long m0 = (long)blockIdx.y * 128, n0 = (long)blockIdx.x * 128;
  const int lr = tid >> 1, lc = (tid & 1) << 4;
  const v8f vz = {0.f, 0.f, 0.f, 0.f, 0.f, 0.f, 0.f, 0.f};
  v8f c[4][2];
#pragma unroll
  for (int mi = 0; mi < 4; ++mi)
#pragma unroll
    for (int ni = 0; ni < 2; ++ni) c[mi][ni] = vz;

  const int kIters = K >> 5;
  const __bf16* ga = A + (m0 + lr) * (long)K + lc;
  const __bf16* gb = W + (n0 + lr) * (long)K + lc;
  for (int kt = 0; kt < kIters; ++kt) {
#ifdef GCTB_HAVE_TDM
    if (tid == 0) {  // TDM: wave-level DMA, EXEC ignored; issue once per block
      gctb_tdm_tile_2d(0u,    A + m0 * (long)K + kt * 32, (unsigned)K, 128u, 128u);
      gctb_tdm_tile_2d(8192u, W + n0 * (long)K + kt * 32, (unsigned)K, 128u, 128u);
      gctb_wait_tensor();
    }
    GCTB_LDS_ESCAPE2(&At[0][0], &Bt[0][0]);
#else
    *(v8bf*)&At[lr][lc]     = *(const v8bf*)(ga);
    *(v8bf*)&At[lr][lc + 8] = *(const v8bf*)(ga + 8);
    *(v8bf*)&Bt[lr][lc]     = *(const v8bf*)(gb);
    *(v8bf*)&Bt[lr][lc + 8] = *(const v8bf*)(gb + 8);
#endif
    ga += 32; gb += 32;
    if (kt + 1 < kIters) {  // pull next K-tile toward L2/WGP$
      __builtin_prefetch(ga, 0, 1);
      __builtin_prefetch(gb, 0, 1);
    }
    __syncthreads();
    v16bf a[4], b[2];
#pragma unroll
    for (int mi = 0; mi < 4; ++mi)
      a[mi] = afrag_bf(&At[wm * 64 + mi * 16 + (lane & 15)][0], lane);
#pragma unroll
    for (int ni = 0; ni < 2; ++ni)
      b[ni] = bfrag_bf(&Bt[wn * 32 + ni * 16 + (lane & 15)][0], lane);
#pragma unroll
    for (int mi = 0; mi < 4; ++mi)
#pragma unroll
      for (int ni = 0; ni < 2; ++ni) c[mi][ni] = wmma_bf16(a[mi], b[ni], c[mi][ni]);
    __syncthreads();
  }
  // C layout: lane l -> col N=l%16 ; VGPR r -> row M = r + (l/16)*8
#pragma unroll
  for (int mi = 0; mi < 4; ++mi) {
#pragma unroll
    for (int ni = 0; ni < 2; ++ni) {
      const long n = n0 + wn * 32 + ni * 16 + (lane & 15);
      const float bv = bias ? bias[n] : 0.f;
#pragma unroll
      for (int r = 0; r < 8; ++r) {
        const long m = m0 + wm * 64 + mi * 16 + ((lane >> 4) << 3) + r;
        const float val = c[mi][ni][r] + bv;
        if (mode == 0) {
          outb[m * (long)outStride + n] = (__bf16)val;
        } else {
          const long t = (m / mapChunk) * mapStride + (m % mapChunk) + mapOff;
          const long idx = t * (long)outStride + n;
          outf[idx] = res[idx] + val;
        }
      }
    }
  }
}

// ---- Fused SwiGLU gate GEMM: H = (A@W1^T+b1) * silu(A@Wg^T+bg), N=1024 ------

__global__ __launch_bounds__(256) void gctb_gemm_gate(
    const __bf16* __restrict__ A, const __bf16* __restrict__ W1b,
    const __bf16* __restrict__ Wgb, const float* __restrict__ b1,
    const float* __restrict__ bg, __bf16* __restrict__ H, int K) {
  // LDS layout (declaration order): At @ 0, B1 @ 8192, Bg @ 16384
  __shared__ __bf16 At[128][32];
  __shared__ __bf16 B1[128][32];
  __shared__ __bf16 Bg[128][32];
  const int tid = threadIdx.x, lane = tid & 31, w = tid >> 5;
  const int wm = w & 1, wn = w >> 1;
  const long m0 = (long)blockIdx.y * 128, n0 = (long)blockIdx.x * 128;
  const int lr = tid >> 1, lc = (tid & 1) << 4;
  const v8f vz = {0.f, 0.f, 0.f, 0.f, 0.f, 0.f, 0.f, 0.f};
  v8f c1[4][2], c2[4][2];
#pragma unroll
  for (int mi = 0; mi < 4; ++mi)
#pragma unroll
    for (int ni = 0; ni < 2; ++ni) { c1[mi][ni] = vz; c2[mi][ni] = vz; }

  const int kIters = K >> 5;
  const __bf16* ga  = A   + (m0 + lr) * (long)K + lc;
  const __bf16* g1  = W1b + (n0 + lr) * (long)K + lc;
  const __bf16* gg  = Wgb + (n0 + lr) * (long)K + lc;
  for (int kt = 0; kt < kIters; ++kt) {
#ifdef GCTB_HAVE_TDM
    if (tid == 0) {
      gctb_tdm_tile_2d(0u,     A   + m0 * (long)K + kt * 32, (unsigned)K, 128u, 128u);
      gctb_tdm_tile_2d(8192u,  W1b + n0 * (long)K + kt * 32, (unsigned)K, 128u, 128u);
      gctb_tdm_tile_2d(16384u, Wgb + n0 * (long)K + kt * 32, (unsigned)K, 128u, 128u);
      gctb_wait_tensor();
    }
    GCTB_LDS_ESCAPE3(&At[0][0], &B1[0][0], &Bg[0][0]);
#else
    *(v8bf*)&At[lr][lc]     = *(const v8bf*)(ga);
    *(v8bf*)&At[lr][lc + 8] = *(const v8bf*)(ga + 8);
    *(v8bf*)&B1[lr][lc]     = *(const v8bf*)(g1);
    *(v8bf*)&B1[lr][lc + 8] = *(const v8bf*)(g1 + 8);
    *(v8bf*)&Bg[lr][lc]     = *(const v8bf*)(gg);
    *(v8bf*)&Bg[lr][lc + 8] = *(const v8bf*)(gg + 8);
#endif
    ga += 32; g1 += 32; gg += 32;
    if (kt + 1 < kIters) {
      __builtin_prefetch(ga, 0, 1);
      __builtin_prefetch(g1, 0, 1);
      __builtin_prefetch(gg, 0, 1);
    }
    __syncthreads();
    v16bf a[4], b1f[2], bgf[2];
#pragma unroll
    for (int mi = 0; mi < 4; ++mi)
      a[mi] = afrag_bf(&At[wm * 64 + mi * 16 + (lane & 15)][0], lane);
#pragma unroll
    for (int ni = 0; ni < 2; ++ni) {
      b1f[ni] = bfrag_bf(&B1[wn * 32 + ni * 16 + (lane & 15)][0], lane);
      bgf[ni] = bfrag_bf(&Bg[wn * 32 + ni * 16 + (lane & 15)][0], lane);
    }
#pragma unroll
    for (int mi = 0; mi < 4; ++mi)
#pragma unroll
      for (int ni = 0; ni < 2; ++ni) {
        c1[mi][ni] = wmma_bf16(a[mi], b1f[ni], c1[mi][ni]);
        c2[mi][ni] = wmma_bf16(a[mi], bgf[ni], c2[mi][ni]);
      }
    __syncthreads();
  }
#pragma unroll
  for (int mi = 0; mi < 4; ++mi) {
#pragma unroll
    for (int ni = 0; ni < 2; ++ni) {
      const long n = n0 + wn * 32 + ni * 16 + (lane & 15);
      const float bv1 = b1[n], bvg = bg[n];
#pragma unroll
      for (int r = 0; r < 8; ++r) {
        const long m = m0 + wm * 64 + mi * 16 + ((lane >> 4) << 3) + r;
        const float u = c1[mi][ni][r] + bv1;
        const float g = c2[mi][ni][r] + bvg;
        const float silu = g / (1.f + __expf(-g));
        H[m * 1024 + n] = (__bf16)(u * silu);
      }
    }
  }
}

// ---- Feature attention: rows of F=32, 8 heads (dh=32), masked softmax -------
// One block per (b,s) row; one wave per head.

__global__ __launch_bounds__(256) void gctb_attn_feat(
    const __bf16* __restrict__ q, const __bf16* __restrict__ k,
    const __bf16* __restrict__ v, const unsigned char* __restrict__ mask,
    __bf16* __restrict__ o) {
  __shared__ float fs[8][32][32];
  const int n = blockIdx.x;                 // 0..B*S-1
  const int h = threadIdx.x >> 5;           // head = wave
  const int lane = threadIdx.x & 31;
  const int b = n >> 9;                     // n / S (S=512)
  const long tok0 = (long)n * 32;
  const __bf16* qh = q + tok0 * 256 + h * 32;
  const __bf16* kh = k + tok0 * 256 + h * 32;
  const __bf16* vh = v + tok0 * 256 + h * 32;
  const float scale = 0.17677669529663687f; // 32^-0.5

  // S = Q @ K^T  (32x32, K=dh=32)
  v16bf aq[2];
#pragma unroll
  for (int mi = 0; mi < 2; ++mi)
    aq[mi] = afrag_bf(qh + (long)(mi * 16 + (lane & 15)) * 256, lane);
#pragma unroll
  for (int ni = 0; ni < 2; ++ni) {
    v16bf bk = bfrag_bf(kh + (long)(ni * 16 + (lane & 15)) * 256, lane);
#pragma unroll
    for (int mi = 0; mi < 2; ++mi) {
      v8f s = {0.f, 0.f, 0.f, 0.f, 0.f, 0.f, 0.f, 0.f};
      s = wmma_bf16(aq[mi], bk, s);
#pragma unroll
      for (int r = 0; r < 8; ++r)
        fs[h][mi * 16 + ((lane >> 4) << 3) + r][ni * 16 + (lane & 15)] = s[r] * scale;
    }
  }
  __syncthreads();

  // masked softmax: lane = query row
  {
    const int fq = lane;
    const unsigned char* mp = mask + ((long)b * 32 + fq) * 32;
    float mx = -INFINITY;
#pragma unroll
    for (int j = 0; j < 32; ++j) {
      float val = mp[j] ? fs[h][fq][j] : -INFINITY;
      mx = fmaxf(mx, val);
    }
    float sum = 0.f;
#pragma unroll
    for (int j = 0; j < 32; ++j) {
      float e = mp[j] ? __expf(fs[h][fq][j] - mx) : 0.f;
      fs[h][fq][j] = e;
      sum += e;
    }
    const float inv = 1.f / sum;
#pragma unroll
    for (int j = 0; j < 32; ++j) fs[h][fq][j] *= inv;
  }
  __syncthreads();

  // O = P @ V  (32x32, K=32 keys)
#pragma unroll
  for (int mi = 0; mi < 2; ++mi) {
    v16bf ap = afrag_f32(&fs[h][mi * 16 + (lane & 15)][0], lane);
#pragma unroll
    for (int ni = 0; ni < 2; ++ni) {
      v16bf bv;
      const int d = ni * 16 + (lane & 15);
      const int gb = (lane >> 4) << 4;
#pragma unroll
      for (int i = 0; i < 16; ++i) bv[i] = vh[(long)(gb + i) * 256 + d];
      v8f oc = {0.f, 0.f, 0.f, 0.f, 0.f, 0.f, 0.f, 0.f};
      oc = wmma_bf16(ap, bv, oc);
      __bf16* oh = o + tok0 * 256 + h * 32;
#pragma unroll
      for (int r = 0; r < 8; ++r) {
        const int f = mi * 16 + ((lane >> 4) << 3) + r;
        oh[(long)f * 256 + d] = (__bf16)oc[r];
      }
    }
  }
}

// ---- Sample attention: one wave per (row n, head h, 32-query tile) ----------
// Scores 32 x Lk (Lk=384) materialized in LDS, two-pass softmax, O via WMMA.

__global__ __launch_bounds__(32) void gctb_attn_samp(
    const __bf16* __restrict__ q, const __bf16* __restrict__ k,
    const __bf16* __restrict__ v, __bf16* __restrict__ o,
    int Lq, int Lk, float scale) {
  __shared__ float fs[32][384];
  const int qt = blockIdx.x, h = blockIdx.y, n = blockIdx.z;
  const int lane = threadIdx.x;
  const __bf16* qh = q + ((long)n * Lq + qt * 32) * 256 + h * 32;
  const __bf16* kh = k + ((long)n * Lk) * 256 + h * 32;
  const __bf16* vh = v + ((long)n * Lk) * 256 + h * 32;

  v16bf aq[2];
#pragma unroll
  for (int mi = 0; mi < 2; ++mi)
    aq[mi] = afrag_bf(qh + (long)(mi * 16 + (lane & 15)) * 256, lane);

  const int kTiles = Lk >> 5;
  for (int kt = 0; kt < kTiles; ++kt) {
#pragma unroll
    for (int ni = 0; ni < 2; ++ni) {
      v16bf bk = bfrag_bf(kh + (long)(kt * 32 + ni * 16 + (lane & 15)) * 256, lane);
#pragma unroll
      for (int mi = 0; mi < 2; ++mi) {
        v8f s = {0.f, 0.f, 0.f, 0.f, 0.f, 0.f, 0.f, 0.f};
        s = wmma_bf16(aq[mi], bk, s);
#pragma unroll
        for (int r = 0; r < 8; ++r)
          fs[mi * 16 + ((lane >> 4) << 3) + r][kt * 32 + ni * 16 + (lane & 15)] = s[r] * scale;
      }
    }
  }
  __syncthreads();  // single-wave workgroup: cheap (S_NOP) but orders LDS

  {  // softmax: lane = query row, over Lk columns
    const int fq = lane;
    float mx = -INFINITY;
    for (int j = 0; j < Lk; ++j) mx = fmaxf(mx, fs[fq][j]);
    float sum = 0.f;
    for (int j = 0; j < Lk; ++j) { float e = __expf(fs[fq][j] - mx); fs[fq][j] = e; sum += e; }
    const float inv = 1.f / sum;
    for (int j = 0; j < Lk; ++j) fs[fq][j] *= inv;
  }
  __syncthreads();

  v8f oc[2][2];
  const v8f vz = {0.f, 0.f, 0.f, 0.f, 0.f, 0.f, 0.f, 0.f};
#pragma unroll
  for (int mi = 0; mi < 2; ++mi)
#pragma unroll
    for (int ni = 0; ni < 2; ++ni) oc[mi][ni] = vz;
  for (int kt = 0; kt < kTiles; ++kt) {
    v16bf ap[2];
#pragma unroll
    for (int mi = 0; mi < 2; ++mi)
      ap[mi] = afrag_f32(&fs[mi * 16 + (lane & 15)][kt * 32], lane);
#pragma unroll
    for (int ni = 0; ni < 2; ++ni) {
      v16bf bv;
      const int d = ni * 16 + (lane & 15);
      const int gb = kt * 32 + ((lane >> 4) << 4);
#pragma unroll
      for (int i = 0; i < 16; ++i) bv[i] = vh[(long)(gb + i) * 256 + d];
#pragma unroll
      for (int mi = 0; mi < 2; ++mi) oc[mi][ni] = wmma_bf16(ap[mi], bv, oc[mi][ni]);
    }
  }
  __bf16* oh = o + ((long)n * Lq + qt * 32) * 256 + h * 32;
#pragma unroll
  for (int mi = 0; mi < 2; ++mi)
#pragma unroll
    for (int ni = 0; ni < 2; ++ni)
#pragma unroll
      for (int r = 0; r < 8; ++r) {
        const int f = mi * 16 + ((lane >> 4) << 3) + r;
        oh[(long)f * 256 + ni * 16 + (lane & 15)] = (__bf16)oc[mi][ni][r];
      }
}

// ---- layout transposes (B,S,F,D) <-> (B,F,S,D), float4 granularity ----------

__global__ __launch_bounds__(256) void gctb_row2col(const float* __restrict__ in,
                                                    float* __restrict__ out) {
  const long i = (long)blockIdx.x * 256 + threadIdx.x;  // over T*D/4
  const long tc = i >> 6;                 // col-token (D/4 = 64 vec4 per token)
  const int qd = (int)(i & 63);
  const long b = tc >> 14, rem = tc & 16383;  // F*S = 16384
  const long f = rem >> 9, s = rem & 511;     // S = 512
  const long tr = (b * 512 + s) * 32 + f;
  ((float4*)out)[tc * 64 + qd] = ((const float4*)in)[tr * 64 + qd];
}

__global__ __launch_bounds__(256) void gctb_col2row(const float* __restrict__ in,
                                                    float* __restrict__ out) {
  const long i = (long)blockIdx.x * 256 + threadIdx.x;
  const long tr = i >> 6;
  const int qd = (int)(i & 63);
  const long b = tr >> 14, rem = tr & 16383;  // S*F = 16384
  const long s = rem >> 5, f = rem & 31;      // F = 32
  const long tc = (b * 32 + f) * 512 + s;
  ((float4*)out)[tr * 64 + qd] = ((const float4*)in)[tc * 64 + qd];
}

// ---- host-side orchestration ------------------------------------------------

extern "C" void kernel_launch(void* const* d_in, const int* in_sizes, int n_in,
                              void* d_out, int out_size, void* d_ws, size_t ws_size,
                              hipStream_t stream) {
  (void)in_sizes; (void)n_in; (void)out_size; (void)ws_size;
  const float* x = (const float*)d_in[0];
  const unsigned char* mask = (const unsigned char*)d_in[1];  // jnp.bool_ -> u8
  // inputs 2/3 are N_train=384 / N_test=128 (fixed by the reference setup)

  const long T = 65536;   // B*S*F tokens
  const long TD = T * 256;
  const long BIG = 1L << 60;
  const float scale = 0.17677669529663687f;  // dh^-0.5, dh=32

  char* base = (char*)d_ws;
  __bf16* wp  = (__bf16*)base;                          // 12x DxD + W1/Wg/W2 bf16
  __bf16* W1b = wp + 12L * 65536;
  __bf16* Wgb = W1b + 262144;
  __bf16* W2b = Wgb + 262144;
  __bf16* xn  = (__bf16*)(base + (4L << 20));           // T*256 bf16
  __bf16* qb  = xn + TD;
  __bf16* kb  = qb + TD;
  __bf16* vb  = kb + TD;
  __bf16* ob  = vb + TD;
  __bf16* hb  = qb;                                     // alias: q/k/v dead at MLP
  float*  xrow = (float*)(ob + TD);                     // residual, row layout
  float*  xcol = xrow + TD;                             // residual, col layout
  float*  outp = (float*)d_out;

  auto Wd = [&](int i) { return wp + (long)i * 65536; };
  // Wd order: 0 Wq_f 1 Wk_f 2 Wv_f 3 Wo_f 4 Wq_t 5 Wk_t 6 Wv_t 7 Wo_t
  //           8 Wq_s 9 Wk_s 10 Wv_s 11 Wo_s

  // 0) weights -> bf16
  {
    const int widx[12] = {4, 5, 6, 7, 12, 13, 14, 15, 20, 21, 22, 23};
    for (int i = 0; i < 12; ++i)
      gctb_cvtw<<<dim3(65536 / 256), 256, 0, stream>>>((const float*)d_in[widx[i]], Wd(i), 65536);
    gctb_cvtw<<<dim3(262144 / 256), 256, 0, stream>>>((const float*)d_in[36], W1b, 262144);
    gctb_cvtw<<<dim3(262144 / 256), 256, 0, stream>>>((const float*)d_in[38], Wgb, 262144);
    gctb_cvtw<<<dim3(262144 / 256), 256, 0, stream>>>((const float*)d_in[40], W2b, 262144);
  }

  // 1) feature attention (row layout), residual from original x
  gctb_ln<<<dim3((unsigned)T), 256, 0, stream>>>(
      x, (const float*)d_in[28], (const float*)d_in[29], xn, BIG, 0, 0);
  dim3 gAll(2, (unsigned)(T / 128));
  gctb_gemm<<<gAll, 256, 0, stream>>>(xn, Wd(0), (const float*)d_in[8],  nullptr, nullptr, qb, 256, 256, BIG, 0, 0, 0);
  gctb_gemm<<<gAll, 256, 0, stream>>>(xn, Wd(1), (const float*)d_in[9],  nullptr, nullptr, kb, 256, 256, BIG, 0, 0, 0);
  gctb_gemm<<<gAll, 256, 0, stream>>>(xn, Wd(2), (const float*)d_in[10], nullptr, nullptr, vb, 256, 256, BIG, 0, 0, 0);
  gctb_attn_feat<<<dim3(2048), 256, 0, stream>>>(qb, kb, vb, mask, ob);
  gctb_gemm<<<gAll, 256, 0, stream>>>(ob, Wd(3), (const float*)d_in[11], x, xrow, nullptr, 256, 256, BIG, 0, 0, 1);
  gctb_row2col<<<dim3(16384), 256, 0, stream>>>(xrow, xcol);

  // 2) train self-attention (col layout, s < 384), in-place residual on xcol
  const long Mtr = 128L * 384, Mte = 128L * 128;
  gctb_ln<<<dim3((unsigned)Mtr), 256, 0, stream>>>(
      xcol, (const float*)d_in[30], (const float*)d_in[31], xn, 384, 512, 0);
  dim3 gTr(2, (unsigned)(Mtr / 128));
  gctb_gemm<<<gTr, 256, 0, stream>>>(xn, Wd(4), (const float*)d_in[16], nullptr, nullptr, qb, 256, 256, BIG, 0, 0, 0);
  gctb_gemm<<<gTr, 256, 0, stream>>>(xn, Wd(5), (const float*)d_in[17], nullptr, nullptr, kb, 256, 256, BIG, 0, 0, 0);
  gctb_gemm<<<gTr, 256, 0, stream>>>(xn, Wd(6), (const float*)d_in[18], nullptr, nullptr, vb, 256, 256, BIG, 0, 0, 0);
  gctb_attn_samp<<<dim3(12, 8, 128), 32, 0, stream>>>(qb, kb, vb, ob, 384, 384, scale);
  gctb_gemm<<<gTr, 256, 0, stream>>>(ob, Wd(7), (const float*)d_in[19], xcol, xcol, nullptr, 256, 256, 384, 512, 0, 1);

  // 3) test -> updated-train cross-attention
  gctb_ln<<<dim3((unsigned)Mtr), 256, 0, stream>>>(       // kv = ln(x_tr, g_s)
      xcol, (const float*)d_in[32], (const float*)d_in[33], xn, 384, 512, 0);
  gctb_ln<<<dim3((unsigned)Mte), 256, 0, stream>>>(       // q  = ln(x_te, g_s)
      xcol, (const float*)d_in[32], (const float*)d_in[33], xn + Mtr * 256, 128, 512, 384);
  dim3 gTe(2, (unsigned)(Mte / 128));
  gctb_gemm<<<gTe, 256, 0, stream>>>(xn + Mtr * 256, Wd(8), (const float*)d_in[24], nullptr, nullptr, qb, 256, 256, BIG, 0, 0, 0);
  gctb_gemm<<<gTr, 256, 0, stream>>>(xn, Wd(9),  (const float*)d_in[25], nullptr, nullptr, kb, 256, 256, BIG, 0, 0, 0);
  gctb_gemm<<<gTr, 256, 0, stream>>>(xn, Wd(10), (const float*)d_in[26], nullptr, nullptr, vb, 256, 256, BIG, 0, 0, 0);
  gctb_attn_samp<<<dim3(4, 8, 128), 32, 0, stream>>>(qb, kb, vb, ob, 128, 384, scale);
  gctb_gemm<<<gTe, 256, 0, stream>>>(ob, Wd(11), (const float*)d_in[27], xcol, xcol, nullptr, 256, 256, 128, 512, 384, 1);

  // 4) back to row layout (xc), SwiGLU MLP, final residual -> d_out
  gctb_col2row<<<dim3(16384), 256, 0, stream>>>(xcol, xrow);
  gctb_ln<<<dim3((unsigned)T), 256, 0, stream>>>(
      xrow, (const float*)d_in[34], (const float*)d_in[35], xn, BIG, 0, 0);
  gctb_gemm_gate<<<dim3(8, (unsigned)(T / 128)), 256, 0, stream>>>(
      xn, W1b, Wgb, (const float*)d_in[37], (const float*)d_in[39], hb, 256);
  gctb_gemm<<<dim3(2, (unsigned)(T / 128)), 256, 0, stream>>>(
      hb, W2b, (const float*)d_in[41], xrow, outp, nullptr, 1024, 256, BIG, 0, 0, 1);
}
